// attentionLayer_14508399526521
// MI455X (gfx1250) — matrix operand order repaired and checked by reading, not compiled, not run on hardware
//
#include <hip/hip_runtime.h>

typedef __attribute__((ext_vector_type(16))) __bf16 v16bf;
typedef __attribute__((ext_vector_type(8)))  __bf16 v8bf;
typedef __attribute__((ext_vector_type(8)))  float  v8f;
typedef __attribute__((ext_vector_type(4)))  float  f4;

#define JX   128
#define JQ   64
#define DD   256

// LDS layout offsets (bytes)
#define OFF_H    0           // 128*256*4 = 131072   h fp32
#define OFF_HW3  131072      // 128*256*2 = 65536    (h*w3) bf16
#define OFF_U    196608      // 64*256*2  = 32768    u bf16 row-major [q][d]
#define OFF_UT   229376      // 256*64*2  = 32768    u bf16 transposed [d][q]
#define OFF_A    262144      // 128*64*2  = 16384    softmax a bf16 [j][q]
#define OFF_W    278528      // 768*4     = 3072
#define OFF_SH   281600      // 128*4
#define OFF_SU   282112      // 64*4
#define OFF_T    282368      // 128*4
#define OFF_P    282880      // 128*4
#define OFF_HA   283392      // 256*4
#define SMEM_BYTES 284416

__device__ __forceinline__ __bf16 f2bf(float f) {
  unsigned u = __builtin_bit_cast(unsigned, f);
  u += 0x7FFFu + ((u >> 16) & 1u);               // round-to-nearest-even
  unsigned short s = (unsigned short)(u >> 16);
  return __builtin_bit_cast(__bf16, s);
}

__device__ __forceinline__ v16bf cat8(v8bf a, v8bf b) {
  return __builtin_shufflevector(a, b, 0,1,2,3,4,5,6,7,8,9,10,11,12,13,14,15);
}

__global__ __launch_bounds__(256)
void bidaf_attn_kernel(const float* __restrict__ H, const float* __restrict__ U,
                       const float* __restrict__ W, const float* __restrict__ Bp,
                       float* __restrict__ Out)
{
  extern __shared__ __align__(32) char smem[];
  float*  sh_h   = (float*) (smem + OFF_H);
  __bf16* sh_hw3 = (__bf16*)(smem + OFF_HW3);
  __bf16* sh_u   = (__bf16*)(smem + OFF_U);
  __bf16* sh_uT  = (__bf16*)(smem + OFF_UT);
  __bf16* sh_a   = (__bf16*)(smem + OFF_A);
  float*  sh_w   = (float*) (smem + OFF_W);
  float*  sh_sh  = (float*) (smem + OFF_SH);
  float*  sh_su  = (float*) (smem + OFF_SU);
  float*  sh_t   = (float*) (smem + OFF_T);
  float*  sh_p   = (float*) (smem + OFF_P);
  float*  sh_ha  = (float*) (smem + OFF_HA);

  const int tid   = threadIdx.x;
  const int wave  = tid >> 5;       // 0..7  (= j-tile)
  const int lane  = tid & 31;
  const int lrow  = lane & 15;      // WMMA B/C/D lane->column, A lane->row
  const int khalf = lane >> 4;      // K-half select per ISA 16-bit layouts

  const int wg = blockIdx.x;        // = b*16 + m
  const int bi = wg >> 4;

  const float* hB = H + (size_t)wg * (JX * DD);
  const float* uB = U + (size_t)bi * (JQ * DD);
  float*       oB = Out + (size_t)wg * (JX * 4 * DD);

  // ---- stage w ----
  for (int i = tid; i < 3 * DD; i += 256) sh_w[i] = W[i];
  __syncthreads();
  const float bb = Bp[0];

  // ---- phase 0: stream h once; build h fp32, (h*w3) bf16, s_h ----
  for (int jr = 0; jr < 16; ++jr) {
    int j  = wave * 16 + jr;
    int d0 = lane * 8;
    f4 x0 = *(const f4*)(hB + (size_t)j * DD + d0);
    f4 x1 = *(const f4*)(hB + (size_t)j * DD + d0 + 4);
    *(f4*)(sh_h + j * DD + d0)     = x0;
    *(f4*)(sh_h + j * DD + d0 + 4) = x1;
    float part = 0.f;
    v8bf hb;
    #pragma unroll
    for (int e = 0; e < 8; ++e) {
      float v = (e < 4) ? x0[e] : x1[e - 4];
      part += v * sh_w[d0 + e];                       // w1
      hb[e] = f2bf(v * sh_w[2 * DD + d0 + e]);        // w3
    }
    *(v8bf*)(sh_hw3 + j * DD + d0) = hb;
    #pragma unroll
    for (int m = 16; m >= 1; m >>= 1) part += __shfl_xor(part, m, 32);
    if (lane == 0) sh_sh[j] = part;
  }
  // ---- phase 0: u -> bf16 (both orientations), s_u ----
  for (int qr = 0; qr < 8; ++qr) {
    int q  = wave * 8 + qr;
    int d0 = lane * 8;
    f4 x0 = *(const f4*)(uB + (size_t)q * DD + d0);
    f4 x1 = *(const f4*)(uB + (size_t)q * DD + d0 + 4);
    float part = 0.f;
    v8bf ub;
    #pragma unroll
    for (int e = 0; e < 8; ++e) {
      float v = (e < 4) ? x0[e] : x1[e - 4];
      part += v * sh_w[DD + d0 + e];                  // w2
      ub[e] = f2bf(v);
    }
    *(v8bf*)(sh_u + q * DD + d0) = ub;
    #pragma unroll
    for (int e = 0; e < 8; ++e) sh_uT[(d0 + e) * JQ + q] = ub[e];
    #pragma unroll
    for (int m = 16; m >= 1; m >>= 1) part += __shfl_xor(part, m, 32);
    if (lane == 0) sh_su[q] = part;
  }
  __syncthreads();

  // ---- GEMM 1: s[16j x 64q] = (h*w3) x u^T, K=256 via 8x WMMA bf16 ----
  const int jrow = wave * 16 + lrow;
  v8f acc[4];
  const v8f vzero = {0.f,0.f,0.f,0.f,0.f,0.f,0.f,0.f};
  #pragma unroll
  for (int qt = 0; qt < 4; ++qt) acc[qt] = vzero;

  #pragma unroll
  for (int ks = 0; ks < 8; ++ks) {
    int kb = ks * 32;
    // A 16x32 bf16 fragment (ISA layout): two 8-elt runs per lane
    v8bf a0 = *(const v8bf*)(sh_hw3 + jrow * DD + kb + khalf * 8);
    v8bf a1 = *(const v8bf*)(sh_hw3 + jrow * DD + kb + 16 + khalf * 8);
    v16bf Af = cat8(a0, a1);
    #pragma unroll
    for (int qt = 0; qt < 4; ++qt) {
      // B 32x16 fragment (ISA layout): two 8-elt K runs of column q
      const __bf16* bp = sh_u + (qt * 16 + lrow) * DD + kb + khalf * 8;
      v16bf Bf = cat8(*(const v8bf*)bp, *(const v8bf*)(bp + 16));
      acc[qt] = __builtin_amdgcn_wmma_f32_16x16x32_bf16(
          false, Af, false, Bf, (short)0, acc[qt], false, false);
    }
  }

  // ---- bias add + row softmax over q (in registers, shfl within 16-lane halves) ----
  #pragma unroll
  for (int r = 0; r < 8; ++r) {
    int j = wave * 16 + khalf * 8 + r;
    float shv = sh_sh[j];
    float mx = -3.402823466e38f;
    #pragma unroll
    for (int qt = 0; qt < 4; ++qt) {
      acc[qt][r] += shv + sh_su[qt * 16 + lrow] + bb;
      mx = fmaxf(mx, acc[qt][r]);
    }
    #pragma unroll
    for (int msk = 8; msk >= 1; msk >>= 1) mx = fmaxf(mx, __shfl_xor(mx, msk, 32));
    if (lrow == 0) sh_t[j] = mx;                      // t[j] = max_q s
    float ssum = 0.f;
    #pragma unroll
    for (int qt = 0; qt < 4; ++qt) {
      float e = expf(acc[qt][r] - mx);
      acc[qt][r] = e; ssum += e;
    }
    #pragma unroll
    for (int msk = 8; msk >= 1; msk >>= 1) ssum += __shfl_xor(ssum, msk, 32);
    float inv = 1.0f / ssum;
    #pragma unroll
    for (int qt = 0; qt < 4; ++qt)
      sh_a[j * JQ + qt * 16 + lrow] = f2bf(acc[qt][r] * inv);
  }
  __syncthreads();

  // ---- p = softmax_j(t) on wave 0 ----
  if (tid < 32) {
    float t0 = sh_t[tid], t1 = sh_t[tid + 32], t2 = sh_t[tid + 64], t3 = sh_t[tid + 96];
    float mx = fmaxf(fmaxf(t0, t1), fmaxf(t2, t3));
    #pragma unroll
    for (int msk = 16; msk >= 1; msk >>= 1) mx = fmaxf(mx, __shfl_xor(mx, msk, 32));
    float e0 = expf(t0 - mx), e1 = expf(t1 - mx), e2 = expf(t2 - mx), e3 = expf(t3 - mx);
    float s = e0 + e1 + e2 + e3;
    #pragma unroll
    for (int msk = 16; msk >= 1; msk >>= 1) s += __shfl_xor(s, msk, 32);
    float inv = 1.0f / s;
    sh_p[tid] = e0 * inv; sh_p[tid + 32] = e1 * inv;
    sh_p[tid + 64] = e2 * inv; sh_p[tid + 96] = e3 * inv;
  }
  __syncthreads();

  // ---- h_att[d] = sum_j p[j] * h[j,d] ----
  {
    float a = 0.f;
    #pragma unroll 4
    for (int j = 0; j < JX; ++j) a += sh_p[j] * sh_h[j * DD + tid];
    sh_ha[tid] = a;
  }
  __syncthreads();

  // ---- GEMM 2: u_att[16j x 256d] = a x u, K=64; fused with output writes ----
  v16bf Aa[2];
  #pragma unroll
  for (int ks = 0; ks < 2; ++ks) {
    int kb = ks * 32;
    v8bf a0 = *(const v8bf*)(sh_a + jrow * JQ + kb + khalf * 8);
    v8bf a1 = *(const v8bf*)(sh_a + jrow * JQ + kb + 16 + khalf * 8);
    Aa[ks] = cat8(a0, a1);
  }
  for (int nt = 0; nt < 16; ++nt) {
    v8f c = vzero;
    #pragma unroll
    for (int ks = 0; ks < 2; ++ks) {
      const __bf16* bp = sh_uT + (nt * 16 + lrow) * JQ + ks * 32 + khalf * 8;
      v16bf Bf = cat8(*(const v8bf*)bp, *(const v8bf*)(bp + 16));
      c = __builtin_amdgcn_wmma_f32_16x16x32_bf16(
          false, Aa[ks], false, Bf, (short)0, c, false, false);
    }
    int dd = nt * 16 + lrow;
    float ha = sh_ha[dd];
    #pragma unroll
    for (int r = 0; r < 8; ++r) {
      int j = wave * 16 + khalf * 8 + r;
      float ua = c[r];
      float hv = sh_h[j * DD + dd];
      float* o = oB + (size_t)j * (4 * DD) + dd;
      o[0]      = hv;          // h
      o[DD]     = ua;          // u_att
      o[2 * DD] = hv * ua;     // h * u_att
      o[3 * DD] = hv * ha;     // h * h_att
    }
  }
}

extern "C" void kernel_launch(void* const* d_in, const int* in_sizes, int n_in,
                              void* d_out, int out_size, void* d_ws, size_t ws_size,
                              hipStream_t stream) {
  (void)in_sizes; (void)n_in; (void)out_size; (void)d_ws; (void)ws_size;
  const float* H  = (const float*)d_in[0];   // (16,16,128,256)
  const float* U  = (const float*)d_in[1];   // (16,64,256)
  const float* W  = (const float*)d_in[2];   // (768,)
  const float* Bb = (const float*)d_in[3];   // scalar
  float* Out = (float*)d_out;                // (16,16,128,1024)
  bidaf_attn_kernel<<<dim3(16 * 16), dim3(256), SMEM_BYTES, stream>>>(H, U, W, Bb, Out);
}
